// TextEncoderMask_66151086293448
// MI455X (gfx1250) — matrix operand order repaired and checked by reading, compiled, not run on hardware
//
#include <hip/hip_runtime.h>
#include <hip/hip_bf16.h>
#include <stdint.h>

typedef __bf16 bf16_t;
typedef __attribute__((ext_vector_type(16))) __bf16 v16bf;
typedef __attribute__((ext_vector_type(8)))  float  v8f;

union FragU { v16bf v; uint32_t u[8]; };

// ---------------------------------------------------------------------------
// WMMA fragment loaders (bf16, 16x16x32). A is MxK row-major; "B" is loaded
// from weights stored [N][K] row-major (i.e. we compute A @ W^T).
// A layout (ISA 7.12.2, 16-bit A 16x32): lane row = lane&15,
//   vgpr i -> K = kk + (i>>2)*16 + (lane>>4)*8 + 2*(i&3)  (+pair)
// B layout (K x N): lane col = lane&15, same K order as A:
//   vgpr i -> K = kk + (i>>2)*16 + (lane>>4)*8 + 2*(i&3)  (+pair)
// Row pointers are hoisted: rp = lane-row dword pointer, loads become
// rp[const + kk/2] so the compiler can strength-reduce addressing.
// ---------------------------------------------------------------------------
__device__ __forceinline__ const uint32_t* lane_row_ptr(const bf16_t* base, int ld,
                                                        int lane) {
  return (const uint32_t*)(base + (size_t)(lane & 15) * ld);
}

__device__ __forceinline__ v16bf load_a_frag(const uint32_t* __restrict__ rp,
                                             int kk, int half) {
  FragU f;
#pragma unroll
  for (int i = 0; i < 8; ++i) {
    int k0 = kk + ((i >> 2) << 4) + (half << 3) + ((i & 3) << 1);
    f.u[i] = rp[k0 >> 1];
  }
  return f.v;
}

__device__ __forceinline__ v16bf load_b_frag(const uint32_t* __restrict__ rp,
                                             int kk, int half) {
  FragU f;
#pragma unroll
  for (int i = 0; i < 8; ++i) {
    int k0 = kk + ((i >> 2) << 4) + (half << 3) + ((i & 3) << 1);
    f.u[i] = rp[k0 >> 1];
  }
  return f.v;
}

__device__ __forceinline__ v8f wmma_bf16(v16bf a, v16bf b, v8f c) {
  return __builtin_amdgcn_wmma_f32_16x16x32_bf16(false, a, false, b, (short)0, c,
                                                 false, false);
}

__device__ __forceinline__ float sigm(float x) { return 1.0f / (1.0f + __expf(-x)); }

// ---------------------------------------------------------------------------
// Generic GEMM: C[M x N] = A[M x K](bf16) @ W[N x K]^T(bf16) + bias[N]
// block = 128 threads (4 waves); block tile = 32 (M) x 256 (N).
// Wave tile = 32 x 64: 2 A-frags x 4 B-frags -> 8 WMMAs per K-step.
// Copy-free ping-pong double buffering; requires K % 64 == 0.
// grid = (M/32, N/256)
// ---------------------------------------------------------------------------
__global__ __launch_bounds__(128, 1)
void gemm_bf16_bias(const bf16_t* __restrict__ A, const bf16_t* __restrict__ W,
                    const float* __restrict__ bias, float* __restrict__ C,
                    int N, int K) {
  const int lane = threadIdx.x & 31;
  const int half = lane >> 4;
  const int wave = threadIdx.x >> 5;
  const int row0 = blockIdx.x << 5;
  const int col0 = (blockIdx.y << 8) + (wave << 6);

  const uint32_t* rpA0 = lane_row_ptr(A + (size_t)row0 * K, K, lane);
  const uint32_t* rpA1 = lane_row_ptr(A + (size_t)(row0 + 16) * K, K, lane);
  const uint32_t* rpB[4];
#pragma unroll
  for (int s = 0; s < 4; ++s)
    rpB[s] = lane_row_ptr(W + (size_t)(col0 + (s << 4)) * K, K, lane);

  v8f acc[2][4];
#pragma unroll
  for (int m = 0; m < 2; ++m)
#pragma unroll
    for (int s = 0; s < 4; ++s)
#pragma unroll
      for (int i = 0; i < 8; ++i) acc[m][s][i] = 0.0f;

  // set X fragments for kk = 0
  v16bf xa0 = load_a_frag(rpA0, 0, half);
  v16bf xa1 = load_a_frag(rpA1, 0, half);
  v16bf xb[4];
#pragma unroll
  for (int s = 0; s < 4; ++s) xb[s] = load_b_frag(rpB[s], 0, half);

  v16bf ya0, ya1, yb[4];
  int kk = 0;
  for (; kk + 64 < K; kk += 64) {
    // load set Y (kk+32) while computing set X (kk)
    ya0 = load_a_frag(rpA0, kk + 32, half);
    ya1 = load_a_frag(rpA1, kk + 32, half);
#pragma unroll
    for (int s = 0; s < 4; ++s) yb[s] = load_b_frag(rpB[s], kk + 32, half);
#pragma unroll
    for (int s = 0; s < 4; ++s) {
      acc[0][s] = wmma_bf16(xa0, xb[s], acc[0][s]);
      acc[1][s] = wmma_bf16(xa1, xb[s], acc[1][s]);
    }
    // load set X (kk+64) while computing set Y (kk+32)
    xa0 = load_a_frag(rpA0, kk + 64, half);
    xa1 = load_a_frag(rpA1, kk + 64, half);
#pragma unroll
    for (int s = 0; s < 4; ++s) xb[s] = load_b_frag(rpB[s], kk + 64, half);
#pragma unroll
    for (int s = 0; s < 4; ++s) {
      acc[0][s] = wmma_bf16(ya0, yb[s], acc[0][s]);
      acc[1][s] = wmma_bf16(ya1, yb[s], acc[1][s]);
    }
  }
  // tail: set X holds kk = K-64; load set Y at K-32, compute both
  ya0 = load_a_frag(rpA0, kk + 32, half);
  ya1 = load_a_frag(rpA1, kk + 32, half);
#pragma unroll
  for (int s = 0; s < 4; ++s) yb[s] = load_b_frag(rpB[s], kk + 32, half);
#pragma unroll
  for (int s = 0; s < 4; ++s) {
    acc[0][s] = wmma_bf16(xa0, xb[s], acc[0][s]);
    acc[1][s] = wmma_bf16(xa1, xb[s], acc[1][s]);
  }
#pragma unroll
  for (int s = 0; s < 4; ++s) {
    acc[0][s] = wmma_bf16(ya0, yb[s], acc[0][s]);
    acc[1][s] = wmma_bf16(ya1, yb[s], acc[1][s]);
  }

  const int nl = lane & 15;
#pragma unroll
  for (int s = 0; s < 4; ++s) {
    int col = col0 + (s << 4) + nl;
    float bv = bias[col];
#pragma unroll
    for (int m = 0; m < 2; ++m) {
#pragma unroll
      for (int r = 0; r < 8; ++r) {
        int row = row0 + (m << 4) + r + (half << 3);
        C[(size_t)row * N + col] = acc[m][s][r] + bv;
      }
    }
  }
}

// ---------------------------------------------------------------------------
// One GRU timestep: gh = h_prev @ W_hh^T + b_hh ; fused gate epilogue.
// block = 128 threads (4 waves); tile = 16 (batch) x 64 (H cols);
// grid = (B/16, H/64). Each wave: 3 accumulators (r,z,n gate columns).
// Copy-free ping-pong K-loop (H % 64 == 0).
// ---------------------------------------------------------------------------
__global__ __launch_bounds__(128, 1)
void gru_step(const bf16_t* __restrict__ h_bf, const float* __restrict__ h_f32,
              const bf16_t* __restrict__ Whh,   // [3H][H] bf16
              const float* __restrict__ bhh,    // [3H]
              const float* __restrict__ xp,     // [B][T][3H]
              const int* __restrict__ lengths,
              float* __restrict__ h_f32_o, bf16_t* __restrict__ h_bf_o,
              float* __restrict__ hs,           // [B][T][H]
              int t, int reverse) {
  const int H = 1024, T = 64, TH = 3072;
  const int lane = threadIdx.x & 31;
  const int half = lane >> 4;
  const int wave = threadIdx.x >> 5;
  const int row0 = blockIdx.x << 4;
  const int col0 = (blockIdx.y << 6) + (wave << 4);

  const uint32_t* rpA = lane_row_ptr(h_bf + (size_t)row0 * H, H, lane);
  const uint32_t* rpB[3];
#pragma unroll
  for (int g = 0; g < 3; ++g)
    rpB[g] = lane_row_ptr(Whh + (size_t)(g * H + col0) * H, H, lane);

  v8f acc[3];
#pragma unroll
  for (int g = 0; g < 3; ++g)
#pragma unroll
    for (int i = 0; i < 8; ++i) acc[g][i] = 0.0f;

  v16bf xa = load_a_frag(rpA, 0, half);
  v16bf xb[3];
#pragma unroll
  for (int g = 0; g < 3; ++g) xb[g] = load_b_frag(rpB[g], 0, half);

  v16bf ya, yb[3];
  int kk = 0;
  for (; kk + 64 < H; kk += 64) {
    ya = load_a_frag(rpA, kk + 32, half);
#pragma unroll
    for (int g = 0; g < 3; ++g) yb[g] = load_b_frag(rpB[g], kk + 32, half);
#pragma unroll
    for (int g = 0; g < 3; ++g) acc[g] = wmma_bf16(xa, xb[g], acc[g]);
    xa = load_a_frag(rpA, kk + 64, half);
#pragma unroll
    for (int g = 0; g < 3; ++g) xb[g] = load_b_frag(rpB[g], kk + 64, half);
#pragma unroll
    for (int g = 0; g < 3; ++g) acc[g] = wmma_bf16(ya, yb[g], acc[g]);
  }
  ya = load_a_frag(rpA, kk + 32, half);
#pragma unroll
  for (int g = 0; g < 3; ++g) yb[g] = load_b_frag(rpB[g], kk + 32, half);
#pragma unroll
  for (int g = 0; g < 3; ++g) acc[g] = wmma_bf16(xa, xb[g], acc[g]);
#pragma unroll
  for (int g = 0; g < 3; ++g) acc[g] = wmma_bf16(ya, yb[g], acc[g]);

  const int col = col0 + (lane & 15);
  const float br = bhh[col], bz = bhh[col + H], bn = bhh[col + 2 * H];
#pragma unroll
  for (int r = 0; r < 8; ++r) {
    int b_ = row0 + r + (half << 3);
    int tt = t;
    if (reverse) { int L = lengths[b_]; tt = (t < L) ? (L - 1 - t) : t; }
    const float* xr = xp + ((size_t)b_ * T + tt) * TH;
    float gr = acc[0][r] + br;
    float gz = acc[1][r] + bz;
    float gn = acc[2][r] + bn;
    float rg = sigm(xr[col] + gr);
    float ug = sigm(xr[col + H] + gz);
    float ng = tanhf(xr[col + 2 * H] + rg * gn);
    float hp = h_f32[(size_t)b_ * H + col];
    float hn = (1.0f - ug) * ng + ug * hp;
    h_f32_o[(size_t)b_ * H + col] = hn;
    h_bf_o[(size_t)b_ * H + col] = (bf16_t)hn;
    hs[((size_t)b_ * T + t) * H + col] = hn;
  }
}

// ---------------------------------------------------------------------------
// One latent-scan step: out = proj[:,t,:] + z_prev @ W_z^T ; reparameterize.
// block = 128 (4 waves); tile = 16 (batch) x 64 (E cols); grid = (B/16, E/64).
// Wave keeps 2 accumulators: mean (out col e) and logv (out col e+E).
// Copy-free ping-pong K-loop (E % 64 == 0).
// ---------------------------------------------------------------------------
__global__ __launch_bounds__(128, 1)
void latent_step(const bf16_t* __restrict__ z_bf,
                 const bf16_t* __restrict__ Wz,    // [2E][E] bf16
                 const float* __restrict__ proj,   // [B][T][2E]
                 const float* __restrict__ eps,    // [B][T][E]
                 float* __restrict__ om, float* __restrict__ ol,
                 float* __restrict__ oz, bf16_t* __restrict__ z_bf_o, int t) {
  const int E = 1024, T = 64, twoE = 2048;
  const int lane = threadIdx.x & 31;
  const int half = lane >> 4;
  const int wave = threadIdx.x >> 5;
  const int row0 = blockIdx.x << 4;
  const int col0 = (blockIdx.y << 6) + (wave << 4);

  const uint32_t* rpA = lane_row_ptr(z_bf + (size_t)row0 * E, E, lane);
  const uint32_t* rpB0 = lane_row_ptr(Wz + (size_t)col0 * E, E, lane);
  const uint32_t* rpB1 = lane_row_ptr(Wz + (size_t)(col0 + E) * E, E, lane);

  v8f acc[2];
#pragma unroll
  for (int g = 0; g < 2; ++g)
#pragma unroll
    for (int i = 0; i < 8; ++i) acc[g][i] = 0.0f;

  v16bf xa  = load_a_frag(rpA, 0, half);
  v16bf xb0 = load_b_frag(rpB0, 0, half);
  v16bf xb1 = load_b_frag(rpB1, 0, half);

  v16bf ya, yb0, yb1;
  int kk = 0;
  for (; kk + 64 < E; kk += 64) {
    ya  = load_a_frag(rpA, kk + 32, half);
    yb0 = load_b_frag(rpB0, kk + 32, half);
    yb1 = load_b_frag(rpB1, kk + 32, half);
    acc[0] = wmma_bf16(xa, xb0, acc[0]);
    acc[1] = wmma_bf16(xa, xb1, acc[1]);
    xa  = load_a_frag(rpA, kk + 64, half);
    xb0 = load_b_frag(rpB0, kk + 64, half);
    xb1 = load_b_frag(rpB1, kk + 64, half);
    acc[0] = wmma_bf16(ya, yb0, acc[0]);
    acc[1] = wmma_bf16(ya, yb1, acc[1]);
  }
  ya  = load_a_frag(rpA, kk + 32, half);
  yb0 = load_b_frag(rpB0, kk + 32, half);
  yb1 = load_b_frag(rpB1, kk + 32, half);
  acc[0] = wmma_bf16(xa, xb0, acc[0]);
  acc[1] = wmma_bf16(xa, xb1, acc[1]);
  acc[0] = wmma_bf16(ya, yb0, acc[0]);
  acc[1] = wmma_bf16(ya, yb1, acc[1]);

  const int col = col0 + (lane & 15);
#pragma unroll
  for (int r = 0; r < 8; ++r) {
    int b_ = row0 + r + (half << 3);
    const float* pr = proj + ((size_t)b_ * T + t) * twoE;
    float mean = pr[col] + acc[0][r];
    float logv = pr[col + E] + acc[1][r];
    float e = eps[((size_t)b_ * T + t) * E + col];
    float z = mean + e * __expf(0.5f * logv);
    size_t oi = ((size_t)b_ * T + t) * E + col;
    om[oi] = mean;
    ol[oi] = logv;
    oz[oi] = z;
    z_bf_o[(size_t)b_ * E + col] = (bf16_t)z;
  }
}

// ---------------------------------------------------------------------------
// Elementwise helpers
// ---------------------------------------------------------------------------
__global__ void f32_to_bf16_k(const float* __restrict__ s, bf16_t* __restrict__ d, int n) {
  int i = blockIdx.x * blockDim.x + threadIdx.x;
  if (i < n) d[i] = (bf16_t)s[i];
}

__global__ void fill_u32_k(uint32_t* __restrict__ p, int n) {
  int i = blockIdx.x * blockDim.x + threadIdx.x;
  if (i < n) p[i] = 0u;
}

__global__ void split_fcw_k(const float* __restrict__ fcW, bf16_t* __restrict__ Wh,
                            bf16_t* __restrict__ Wz, int twoE, int twoH, int E) {
  int i = blockIdx.x * blockDim.x + threadIdx.x;
  int cols = twoH + E;
  if (i >= twoE * cols) return;
  int o = i / cols, c = i % cols;
  float v = fcW[i];
  if (c < twoH) Wh[(size_t)o * twoH + c] = (bf16_t)v;
  else          Wz[(size_t)o * E + (c - twoH)] = (bf16_t)v;
}

__global__ void embed_gather_k(const int* __restrict__ x, const float* __restrict__ emb,
                               bf16_t* __restrict__ xe, int BT, int D) {
  int i = blockIdx.x * blockDim.x + threadIdx.x;
  if (i >= BT * D) return;
  int bt = i / D, d = i % D;
  xe[i] = (bf16_t)emb[(size_t)x[bt] * D + d];
}

// mask + un-reverse hb + build bf16 hcat + length-normalized hidden_o
__global__ void hcat_hidden_k(const float* __restrict__ hf,
                              const float* __restrict__ hb_rev,
                              const int* __restrict__ lengths,
                              bf16_t* __restrict__ hcat,   // [B][T][2H]
                              float* __restrict__ hidden_o) {
  const int H = 1024, T = 64, twoH = 2048;
  int idx = blockIdx.x * blockDim.x + threadIdx.x;  // B*H
  int b = idx / H, j = idx % H;
  int L = lengths[b];
  float inv_len = 1.0f / (float)L;
  float sf = 0.0f, sb = 0.0f;
  for (int t = 0; t < T; ++t) {
    float m = (t < L) ? 1.0f : 0.0f;
    float fv = hf[((size_t)b * T + t) * H + j] * m;
    int tt = (t < L) ? (L - 1 - t) : t;
    float bv = hb_rev[((size_t)b * T + tt) * H + j] * m;
    sf += fv; sb += bv;
    hcat[((size_t)b * T + t) * twoH + j]     = (bf16_t)fv;
    hcat[((size_t)b * T + t) * twoH + H + j] = (bf16_t)bv;
  }
  hidden_o[idx] = (sf + sb) * inv_len * 0.5f;
}

// ---------------------------------------------------------------------------
// Host orchestration
// ---------------------------------------------------------------------------
extern "C" void kernel_launch(void* const* d_in, const int* in_sizes, int n_in,
                              void* d_out, int out_size, void* d_ws, size_t ws_size,
                              hipStream_t stream) {
  (void)in_sizes; (void)n_in; (void)out_size; (void)ws_size;
  const int B = 256, T = 64, D = 512, H = 1024, E = 1024;
  const int TH = 3 * H, twoH = 2 * H, twoE = 2 * E, BT = B * T;

  const int*   x       = (const int*)d_in[0];
  const int*   lengths = (const int*)d_in[1];
  const float* emb     = (const float*)d_in[2];
  const float* W_ih_f  = (const float*)d_in[3];
  const float* W_hh_f  = (const float*)d_in[4];
  const float* b_ih_f  = (const float*)d_in[5];
  const float* b_hh_f  = (const float*)d_in[6];
  const float* W_ih_b  = (const float*)d_in[7];
  const float* W_hh_b  = (const float*)d_in[8];
  const float* b_ih_b  = (const float*)d_in[9];
  const float* b_hh_b  = (const float*)d_in[10];
  const float* fc_W    = (const float*)d_in[11];
  const float* fc_b    = (const float*)d_in[12];
  const float* eps     = (const float*)d_in[13];

  float* out_means  = (float*)d_out;
  float* out_logs   = out_means + (size_t)BT * E;
  float* out_z      = out_logs  + (size_t)BT * E;
  float* out_hidden = out_z     + (size_t)BT * E;

  char* wp = (char*)d_ws;
  auto alloc = [&](size_t bytes) -> void* {
    void* p = (void*)wp;
    wp += (bytes + 255) & ~(size_t)255;
    return p;
  };
  bf16_t* xe_bf   = (bf16_t*)alloc((size_t)BT * D * 2);
  bf16_t* Wihf_bf = (bf16_t*)alloc((size_t)TH * D * 2);
  bf16_t* Wihb_bf = (bf16_t*)alloc((size_t)TH * D * 2);
  bf16_t* Whhf_bf = (bf16_t*)alloc((size_t)TH * H * 2);
  bf16_t* Whhb_bf = (bf16_t*)alloc((size_t)TH * H * 2);
  bf16_t* Wh_bf   = (bf16_t*)alloc((size_t)twoE * twoH * 2);
  bf16_t* Wz_bf   = (bf16_t*)alloc((size_t)twoE * E * 2);
  float*  xp_f    = (float*)alloc((size_t)BT * TH * 4);
  float*  xp_b    = (float*)alloc((size_t)BT * TH * 4);
  float*  hf      = (float*)alloc((size_t)BT * H * 4);
  float*  hb_rev  = (float*)alloc((size_t)BT * H * 4);
  float*  h32a    = (float*)alloc((size_t)B * H * 4);
  float*  h32b    = (float*)alloc((size_t)B * H * 4);
  bf16_t* hbfa    = (bf16_t*)alloc((size_t)B * H * 2);
  bf16_t* hbfb    = (bf16_t*)alloc((size_t)B * H * 2);
  bf16_t* hcat    = (bf16_t*)alloc((size_t)BT * twoH * 2);
  float*  proj    = (float*)alloc((size_t)BT * twoE * 4);
  bf16_t* zbfa    = (bf16_t*)alloc((size_t)B * E * 2);
  bf16_t* zbfb    = (bf16_t*)alloc((size_t)B * E * 2);

  const int TPB = 256;
  auto blks = [](int n, int t) { return (n + t - 1) / t; };

  // 1) weight conversions + embedding gather
  f32_to_bf16_k<<<blks(TH * D, TPB), TPB, 0, stream>>>(W_ih_f, Wihf_bf, TH * D);
  f32_to_bf16_k<<<blks(TH * D, TPB), TPB, 0, stream>>>(W_ih_b, Wihb_bf, TH * D);
  f32_to_bf16_k<<<blks(TH * H, TPB), TPB, 0, stream>>>(W_hh_f, Whhf_bf, TH * H);
  f32_to_bf16_k<<<blks(TH * H, TPB), TPB, 0, stream>>>(W_hh_b, Whhb_bf, TH * H);
  split_fcw_k<<<blks(twoE * (twoH + E), TPB), TPB, 0, stream>>>(fc_W, Wh_bf, Wz_bf,
                                                               twoE, twoH, E);
  embed_gather_k<<<blks(BT * D, TPB), TPB, 0, stream>>>(x, emb, xe_bf, BT, D);

  // 2) input projections (WMMA GEMM): xp = xe @ W_ih^T + b_ih
  gemm_bf16_bias<<<dim3(BT / 32, TH / 256), 128, 0, stream>>>(xe_bf, Wihf_bf, b_ih_f,
                                                              xp_f, TH, D);
  gemm_bf16_bias<<<dim3(BT / 32, TH / 256), 128, 0, stream>>>(xe_bf, Wihb_bf, b_ih_b,
                                                              xp_b, TH, D);

  // 3) forward GRU scan
  fill_u32_k<<<blks(B * H, TPB), TPB, 0, stream>>>((uint32_t*)h32a, B * H);
  fill_u32_k<<<blks(B * H / 2, TPB), TPB, 0, stream>>>((uint32_t*)hbfa, B * H / 2);
  {
    float* hin32 = h32a; float* hout32 = h32b;
    bf16_t* hinb = hbfa; bf16_t* houtb = hbfb;
    for (int t = 0; t < T; ++t) {
      gru_step<<<dim3(B / 16, H / 64), 128, 0, stream>>>(hinb, hin32, Whhf_bf, b_hh_f,
                                                         xp_f, lengths, hout32, houtb,
                                                         hf, t, 0);
      float* t32 = hin32; hin32 = hout32; hout32 = t32;
      bf16_t* tb = hinb; hinb = houtb; houtb = tb;
    }
  }

  // 4) backward GRU scan (time-reversed input indexing, output stored reversed)
  fill_u32_k<<<blks(B * H, TPB), TPB, 0, stream>>>((uint32_t*)h32a, B * H);
  fill_u32_k<<<blks(B * H / 2, TPB), TPB, 0, stream>>>((uint32_t*)hbfa, B * H / 2);
  {
    float* hin32 = h32a; float* hout32 = h32b;
    bf16_t* hinb = hbfa; bf16_t* houtb = hbfb;
    for (int t = 0; t < T; ++t) {
      gru_step<<<dim3(B / 16, H / 64), 128, 0, stream>>>(hinb, hin32, Whhb_bf, b_hh_b,
                                                         xp_b, lengths, hout32, houtb,
                                                         hb_rev, t, 1);
      float* t32 = hin32; hin32 = hout32; hout32 = t32;
      bf16_t* tb = hinb; hinb = houtb; houtb = tb;
    }
  }

  // 5) mask + unreverse + hcat assembly + hidden_o reduction
  hcat_hidden_k<<<blks(B * H, TPB), TPB, 0, stream>>>(hf, hb_rev, lengths, hcat,
                                                      out_hidden);

  // 6) proj = hcat @ W_h^T + fc_b   (WMMA GEMM)
  gemm_bf16_bias<<<dim3(BT / 32, twoE / 256), 128, 0, stream>>>(hcat, Wh_bf, fc_b,
                                                                proj, twoE, twoH);

  // 7) latent autoregressive scan
  fill_u32_k<<<blks(B * E / 2, TPB), TPB, 0, stream>>>((uint32_t*)zbfa, B * E / 2);
  {
    bf16_t* zin = zbfa; bf16_t* zout = zbfb;
    for (int t = 0; t < T; ++t) {
      latent_step<<<dim3(B / 16, E / 64), 128, 0, stream>>>(zin, Wz_bf, proj, eps,
                                                            out_means, out_logs, out_z,
                                                            zout, t);
      bf16_t* tz = zin; zin = zout; zout = tz;
    }
  }
}
